// newHGkNN_80350248174049
// MI455X (gfx1250) — matrix operand order repaired and checked by reading, not compiled, not run on hardware
//
#include <hip/hip_runtime.h>
#include <hip/hip_bf16.h>

#define B_SZ 32
#define NX 16384
#define IN_DIM 3
#define DIM 64
#define JJ 16
#define MODE 128
#define RANK 4
#define FC_DIM 128
#define N_LAYERS 3

typedef __attribute__((ext_vector_type(16))) __bf16 v16bf;
typedef __attribute__((ext_vector_type(8)))  __bf16 v8bf;
typedef __attribute__((ext_vector_type(8)))  float  v8f;
typedef __attribute__((ext_vector_type(4)))  unsigned v4u;
typedef __attribute__((ext_vector_type(8)))  int      v8i_t;
typedef __attribute__((ext_vector_type(4)))  int      v4i_t;

__device__ __forceinline__ __bf16 f2bf(float f) {
  union { float f; unsigned u; } v; v.f = f;
  unsigned r = (v.u + 0x7FFFu + ((v.u >> 16) & 1u)) >> 16;
  union { unsigned short s; __bf16 b; } o; o.s = (unsigned short)r;
  return o.b;
}

__device__ __forceinline__ v8f wmma_bf(v16bf a, v16bf b, v8f c) {
  return __builtin_amdgcn_wmma_f32_16x16x32_bf16(false, a, false, b, (short)0, c,
                                                 false, false);
}

// ---- WMMA operand loaders (ISA 7.12.2 layouts, wave32), pure bf16 vector loads ----
// A 16x32 (MxK): lane m=l&15, half=l>>4; K chunks [8h..8h+7] and [16+8h..16+8h+7]
__device__ __forceinline__ v16bf load_a_bf(const __bf16* A, int lda) {
  int lane = threadIdx.x & 31;
  const __bf16* p = A + (size_t)(lane & 15) * lda + ((lane >> 4) << 3);
  v8bf lo = *(const v8bf*)p;
  v8bf hi = *(const v8bf*)(p + 16);
  return __builtin_shufflevector(lo, hi, 0, 1, 2, 3, 4, 5, 6, 7, 8, 9, 10, 11, 12,
                                 13, 14, 15);
}
// B 32x16 (KxN): lane n=l&15, half=l>>4; K rows [8h..8h+7] and [16+8h..16+8h+7]
__device__ __forceinline__ v16bf load_b_bf(const __bf16* B, int ldb) {
  int lane = threadIdx.x & 31;
  const __bf16* p = B + (lane & 15) + (size_t)((lane >> 4) << 3) * ldb;
  v16bf r;
#pragma unroll
  for (int s = 0; s < 8; ++s) {
    r[s] = p[(size_t)s * ldb];
    r[8 + s] = p[(size_t)(16 + s) * ldb];
  }
  return r;
}

__device__ __forceinline__ float gelu_f(float x) {
  return 0.5f * x * (1.0f + erff(x * 0.7071067811865475f));
}

// ---- Tensor Data Mover: DMA a 128-row x 64-col bf16 panel (row stride MODE) to LDS ----
// D# per CDNA5 ISA ch.8: group0 {count=1, lds_addr, global_addr, type=2},
// group1 {data_size=1(2B), tensor_dim0=64, tensor_dim1=128, tile_dim0=64,
//         tile_dim1=128, tensor_dim0_stride=128}; groups 2/3 zero (2-D tile).
__device__ __forceinline__ void tdm_load_panel(const __bf16* gsrc, __bf16* lds_dst) {
  unsigned lds_off = (unsigned)(unsigned long long)lds_dst;  // generic LDS addr[31:0]
  unsigned long long ga = (unsigned long long)gsrc;
  v4u g0 = {1u,                                   // count=1, is_restore=0
            lds_off,                              // lds_addr
            (unsigned)ga,                         // global_addr[31:0] -> bits 95:64
            ((unsigned)(ga >> 32) & 0x01FFFFFFu)  // global_addr[56:32] -> bits 120:96
                | 0x80000000u};                   // type=2 -> bits 127:126
  v8i_t g1 = {(int)(1u << 16),    // data_size = 1 (2 bytes); mask=0
              (int)(64u << 16),   // tensor_dim0 = 64 (low16 -> bits 63:48)
              (int)(128u << 16),  // tensor_dim0 hi=0; tensor_dim1 = 128 (low16)
              (int)(64u << 16),   // tensor_dim1 hi=0; tile_dim0 = 64
              128,                // tile_dim1 = 128; tile_dim2 = 0
              128,                // tensor_dim0_stride = 128 (low32)
              0,                  // stride0 hi; tensor_dim1_stride low (unused, 2-D)
              0};
  v4i_t z4 = {0, 0, 0, 0};
  v8i_t z8 = {0, 0, 0, 0, 0, 0, 0, 0};
  __builtin_amdgcn_tensor_load_to_lds(g0, g1, z4, z4, z8, 0);
}

// ---------------- precompute kernels (one-time bf16 mirrors) ----------------
__global__ void k_prep_H(const float* Do, const float* Di, const float* pr,
                         const float* A, const float* Bm, unsigned short* HT_) {
  int t = blockIdx.x * blockDim.x + threadIdx.x;
  if (t >= JJ * MODE * MODE) return;
  int k = t & (MODE - 1);
  int l = (t >> 7) & (MODE - 1);
  int j = t >> 14;
  float v = Do[j * MODE + k] * Di[j * MODE + l] * pr[k * MODE + l];
#pragma unroll
  for (int r = 0; r < RANK; ++r)
    v += A[(j * RANK + r) * MODE + k] * Bm[(j * RANK + r) * MODE + l];
  ((__bf16*)HT_)[(j * MODE + l) * MODE + k] = f2bf(v);
}

__global__ void k_prep_wsp(const float* Wsp, unsigned short* WT_) {
  int t = blockIdx.x * blockDim.x + threadIdx.x;
  if (t >= N_LAYERS * JJ * DIM * DIM) return;
  int i = t & 63;
  int o = (t >> 6) & 63;
  int j = (t >> 12) & 15;
  int lay = t >> 16;
  ((__bf16*)WT_)[t] = f2bf(Wsp[((lay * DIM + i) * DIM + o) * JJ + j]);
}

__global__ void k_prep_basesT(const float* bases, unsigned short* BT_) {
  int t = blockIdx.x * blockDim.x + threadIdx.x;
  if (t >= MODE * NX) return;
  int xi = t & (NX - 1);
  int l = t >> 14;
  ((__bf16*)BT_)[t] = f2bf(bases[(size_t)xi * MODE + l]);
}

__global__ void k_prep_cvt(const float* src, unsigned short* dst, int n) {
  int t = blockIdx.x * blockDim.x + threadIdx.x;
  if (t >= n) return;
  ((__bf16*)dst)[t] = f2bf(src[t]);
}

__global__ void k_prep_w1t(const float* W1, unsigned short* WT_) {
  int t = blockIdx.x * blockDim.x + threadIdx.x;
  if (t >= FC_DIM * DIM) return;
  int c = t & 63;
  int f = t >> 6;
  ((__bf16*)WT_)[t] = f2bf(W1[c * FC_DIM + f]);
}

// ---------------- fc0: h[b][c][x] = x[b][x][:]·W0[:,c] + b0[c]  (bf16 out) ----------------
__global__ void k_fc0(const float* x, const float* W0, const float* b0,
                      unsigned short* h_) {
  __bf16* h = (__bf16*)h_;
  int t = blockIdx.x * blockDim.x + threadIdx.x;  // over B*NX
  int xi = t & (NX - 1);
  int b = t >> 14;
  float x0 = x[(size_t)t * 3 + 0];
  float x1 = x[(size_t)t * 3 + 1];
  float x2 = x[(size_t)t * 3 + 2];
#pragma unroll 4
  for (int c = 0; c < DIM; ++c) {
    float v = x0 * W0[c] + x1 * W0[DIM + c] + x2 * W0[2 * DIM + c] + b0[c];
    h[((size_t)b * DIM + c) * NX + xi] = f2bf(v);
  }
}

// ---------------- xh = h · wbases   (M=2048, K=16384, N=128, all bf16) ----------------
// block 128 thr = 4 waves; wave: 32 rows x 64 cols. B panel (shared by all 4 waves)
// staged to LDS by the Tensor Data Mover, double-buffered on TENSORcnt.
__global__ void __launch_bounds__(128) k_gemm_xh(const unsigned short* h_,
                                                 const unsigned short* wb_,
                                                 unsigned short* xh_) {
  __shared__ __bf16 s_B[2][128 * 64];  // 2 x 16 KB
  const __bf16* h = (const __bf16*)h_;
  const __bf16* wb = (const __bf16*)wb_;
  __bf16* xh = (__bf16*)xh_;
  int wave = threadIdx.x >> 5;
  int row0 = blockIdx.y * 128 + wave * 32;
  int col0 = blockIdx.x * 64;
  const __bf16* A0 = h + (size_t)row0 * NX;
  const __bf16* A1 = h + (size_t)(row0 + 16) * NX;
  v8f acc[2][4] = {};
  if (wave == 0) tdm_load_panel(wb + col0, s_B[0]);
  const int NCH = NX / 128;
  for (int ch = 0; ch < NCH; ++ch) {
    if (wave == 0) {
      __builtin_amdgcn_s_wait_tensorcnt(0);  // current buffer resident
      if (ch + 1 < NCH)
        tdm_load_panel(wb + (size_t)(ch + 1) * 128 * MODE + col0, s_B[(ch + 1) & 1]);
    }
    __syncthreads();  // publish buffer to all waves; DMA of next overlaps compute
    const __bf16* Bp = s_B[ch & 1];
    int kbase = ch * 128;
#pragma unroll
    for (int s = 0; s < 4; ++s) {
      int k0 = kbase + s * 32;
      v16bf a0 = load_a_bf(A0 + k0, NX);
      v16bf a1 = load_a_bf(A1 + k0, NX);
#pragma unroll
      for (int t = 0; t < 4; ++t) {
        v16bf b = load_b_bf(Bp + (s * 32) * 64 + t * 16, 64);
        acc[0][t] = wmma_bf(a0, b, acc[0][t]);
        acc[1][t] = wmma_bf(a1, b, acc[1][t]);
      }
    }
    __syncthreads();  // all waves done with this buffer before it is overwritten
  }
  int lane = threadIdx.x & 31, n = lane & 15, half = lane >> 4;
#pragma unroll
  for (int s = 0; s < 2; ++s)
#pragma unroll
    for (int t = 0; t < 4; ++t)
#pragma unroll
      for (int r = 0; r < 8; ++r)
        xh[(row0 + s * 16 + r + 8 * half) * MODE + col0 + t * 16 + n] =
            f2bf(acc[s][t][r]);
}

// ---------------- spectral mixing: y[b] = sum_j (Wsp^T[j]·xh[b])·H^T[j] ----------------
__global__ void __launch_bounds__(256) k_mid(const unsigned short* xh_,
                                             const unsigned short* WT_,
                                             const unsigned short* HT_,
                                             unsigned short* y_) {
  __shared__ __bf16 s_xhb[DIM * MODE];   // 16 KB
  __shared__ __bf16 s_G[8 * 16 * MODE];  // 32 KB (wave-private slices)
  const __bf16* WT = (const __bf16*)WT_;
  const __bf16* HT = (const __bf16*)HT_;
  __bf16* y = (__bf16*)y_;
  int b = blockIdx.x;
  {
    const uint4* src = (const uint4*)(xh_ + (size_t)b * DIM * MODE);
    uint4* dst = (uint4*)s_xhb;
    for (int t = threadIdx.x; t < DIM * MODE / 8; t += 256) dst[t] = src[t];
  }
  __syncthreads();
  int wave = threadIdx.x >> 5;
  int orow = (wave & 3) * 16;
  int kcol = (wave >> 2) * 64;
  __bf16* myG = s_G + wave * (16 * MODE);
  int lane = threadIdx.x & 31, n = lane & 15, half = lane >> 4;
  v8f accY[4] = {};
  for (int j = 0; j < JJ; ++j) {
    v8f g[8] = {};
    const __bf16* Wj = WT + (j * DIM + orow) * DIM;
#pragma unroll
    for (int kk = 0; kk < DIM; kk += 32) {
      v16bf a = load_a_bf(Wj + kk, DIM);
#pragma unroll
      for (int t = 0; t < 8; ++t) {
        v16bf bb = load_b_bf(s_xhb + kk * MODE + t * 16, MODE);
        g[t] = wmma_bf(a, bb, g[t]);
      }
    }
#pragma unroll
    for (int t = 0; t < 8; ++t)
#pragma unroll
      for (int r = 0; r < 8; ++r)
        myG[(r + 8 * half) * MODE + t * 16 + n] = f2bf(g[t][r]);
    const __bf16* Hj = HT + j * MODE * MODE + kcol;
#pragma unroll
    for (int kk = 0; kk < MODE; kk += 32) {
      v16bf a = load_a_bf(myG + kk, MODE);
#pragma unroll
      for (int t = 0; t < 4; ++t)
        accY[t] = wmma_bf(a, load_b_bf(Hj + kk * MODE + t * 16, MODE), accY[t]);
    }
  }
#pragma unroll
  for (int t = 0; t < 4; ++t)
#pragma unroll
    for (int r = 0; r < 8; ++r)
      y[b * DIM * MODE + (orow + r + 8 * half) * MODE + kcol + t * 16 + n] =
          f2bf(accY[t][r]);
}

// ---------------- h = gelu?( y·bases^T + W_conv·h + b_conv )  (in-place bf16) ----------------
__global__ void __launch_bounds__(256) k_ispec(const unsigned short* y_,
                                               const unsigned short* BT_,
                                               const unsigned short* Wc_,
                                               const float* bconv,
                                               unsigned short* h_, int do_gelu) {
  const __bf16* y = (const __bf16*)y_;
  const __bf16* BT = (const __bf16*)BT_;
  const __bf16* Wc = (const __bf16*)Wc_;
  __bf16* h = (__bf16*)h_;
  int b = blockIdx.y;
  int wave = threadIdx.x >> 5;
  int orow = (wave & 3) * 16;
  int x0 = blockIdx.x * 128 + (wave >> 2) * 64;
  v8f acc[4] = {};
  const __bf16* Ay = y + b * DIM * MODE + orow * MODE;
#pragma unroll
  for (int kk = 0; kk < MODE; kk += 32) {
    v16bf a = load_a_bf(Ay + kk, MODE);
#pragma unroll
    for (int t = 0; t < 4; ++t)
      acc[t] = wmma_bf(a, load_b_bf(BT + (size_t)kk * NX + x0 + t * 16, NX), acc[t]);
  }
  const __bf16* Bh = h + (size_t)b * DIM * NX;
#pragma unroll
  for (int kk = 0; kk < DIM; kk += 32) {
    v16bf a = load_a_bf(Wc + orow * DIM + kk, DIM);
#pragma unroll
    for (int t = 0; t < 4; ++t)
      acc[t] = wmma_bf(a, load_b_bf(Bh + (size_t)kk * NX + x0 + t * 16, NX), acc[t]);
  }
  __syncthreads();  // all waves done reading h[b][:, x-tile] before in-place write
  int lane = threadIdx.x & 31, n = lane & 15, half = lane >> 4;
#pragma unroll
  for (int t = 0; t < 4; ++t)
#pragma unroll
    for (int r = 0; r < 8; ++r) {
      int o = orow + r + 8 * half;
      float v = acc[t][r] + bconv[o];
      if (do_gelu) v = gelu_f(v);
      h[((size_t)b * DIM + o) * NX + x0 + t * 16 + n] = f2bf(v);
    }
}

// ---------------- out = gelu(h^T·W1 + b1)·W2 + b2 ----------------
__global__ void __launch_bounds__(256) k_final(const unsigned short* h_,
                                               const unsigned short* W1T_,
                                               const float* b1, const float* W2,
                                               const float* b2, float* out) {
  const __bf16* h = (const __bf16*)h_;
  const __bf16* W1T = (const __bf16*)W1T_;
  __shared__ float s_acc[64];
  int b = blockIdx.y;
  int x0 = blockIdx.x * 64;
  int wave = threadIdx.x >> 5;
  int frow = wave * 16;
  v8f acc[4] = {};
  const __bf16* Bh = h + (size_t)b * DIM * NX;
#pragma unroll
  for (int kk = 0; kk < DIM; kk += 32) {
    v16bf a = load_a_bf(W1T + frow * DIM + kk, DIM);
#pragma unroll
    for (int t = 0; t < 4; ++t)
      acc[t] = wmma_bf(a, load_b_bf(Bh + (size_t)kk * NX + x0 + t * 16, NX), acc[t]);
  }
  if (threadIdx.x < 64) s_acc[threadIdx.x] = 0.f;
  __syncthreads();
  int lane = threadIdx.x & 31, n = lane & 15, half = lane >> 4;
#pragma unroll
  for (int t = 0; t < 4; ++t) {
    float partial = 0.f;
#pragma unroll
    for (int r = 0; r < 8; ++r) {
      int f = frow + r + 8 * half;
      partial += gelu_f(acc[t][r] + b1[f]) * W2[f];
    }
    atomicAdd(&s_acc[t * 16 + n], partial);
  }
  __syncthreads();
  if (threadIdx.x < 64)
    out[(size_t)b * NX + x0 + threadIdx.x] = s_acc[threadIdx.x] + b2[0];
}

extern "C" void kernel_launch(void* const* d_in, const int* in_sizes, int n_in,
                              void* d_out, int out_size, void* d_ws, size_t ws_size,
                              hipStream_t stream) {
  const float* x       = (const float*)d_in[0];
  const float* bases   = (const float*)d_in[1];
  const float* wbases  = (const float*)d_in[2];
  const float* product = (const float*)d_in[3];
  const float* D_out   = (const float*)d_in[4];
  const float* D_in    = (const float*)d_in[5];
  const float* A       = (const float*)d_in[6];
  const float* Bm      = (const float*)d_in[7];
  const float* W_sp    = (const float*)d_in[8];
  const float* W_conv  = (const float*)d_in[9];
  const float* b_conv  = (const float*)d_in[10];
  const float* W0      = (const float*)d_in[11];
  const float* b0      = (const float*)d_in[12];
  const float* W1      = (const float*)d_in[13];
  const float* b1      = (const float*)d_in[14];
  const float* W2      = (const float*)d_in[15];
  const float* b2      = (const float*)d_in[16];
  float* out = (float*)d_out;

  char* w = (char*)d_ws;
  unsigned short* h   = (unsigned short*)w; w += (size_t)B_SZ * DIM * NX * 2;  // 67 MB
  unsigned short* xh  = (unsigned short*)w; w += (size_t)B_SZ * DIM * MODE * 2;
  unsigned short* yb  = (unsigned short*)w; w += (size_t)B_SZ * DIM * MODE * 2;
  unsigned short* HT  = (unsigned short*)w; w += (size_t)JJ * MODE * MODE * 2;
  unsigned short* WT  = (unsigned short*)w; w += (size_t)N_LAYERS * JJ * DIM * DIM * 2;
  unsigned short* BT  = (unsigned short*)w; w += (size_t)MODE * NX * 2;
  unsigned short* WB  = (unsigned short*)w; w += (size_t)NX * MODE * 2;  // wbases bf16
  unsigned short* WCB = (unsigned short*)w; w += (size_t)N_LAYERS * DIM * DIM * 2;
  unsigned short* W1T = (unsigned short*)w; w += (size_t)FC_DIM * DIM * 2;

  k_prep_H<<<(JJ * MODE * MODE) / 256, 256, 0, stream>>>(D_out, D_in, product, A, Bm, HT);
  k_prep_wsp<<<(N_LAYERS * JJ * DIM * DIM) / 256, 256, 0, stream>>>(W_sp, WT);
  k_prep_basesT<<<(MODE * NX) / 256, 256, 0, stream>>>(bases, BT);
  k_prep_cvt<<<(NX * MODE) / 256, 256, 0, stream>>>(wbases, WB, NX * MODE);
  k_prep_cvt<<<(N_LAYERS * DIM * DIM + 255) / 256, 256, 0, stream>>>(
      W_conv, WCB, N_LAYERS * DIM * DIM);
  k_prep_w1t<<<(FC_DIM * DIM) / 256, 256, 0, stream>>>(W1, W1T);
  k_fc0<<<(B_SZ * NX) / 256, 256, 0, stream>>>(x, W0, b0, h);

  for (int lay = 0; lay < N_LAYERS; ++lay) {
    k_gemm_xh<<<dim3(MODE / 64, (B_SZ * DIM) / 128), 128, 0, stream>>>(h, WB, xh);
    k_mid<<<B_SZ, 256, 0, stream>>>(xh, WT + (size_t)lay * JJ * DIM * DIM, HT, yb);
    k_ispec<<<dim3(NX / 128, B_SZ), 256, 0, stream>>>(
        yb, BT, WCB + (size_t)lay * DIM * DIM, b_conv + lay * DIM, h,
        (lay < N_LAYERS - 1) ? 1 : 0);
  }
  k_final<<<dim3(NX / 64, B_SZ), 256, 0, stream>>>(h, W1T, b1, W2, b2, out);
}
